// WReN_31078383354418
// MI455X (gfx1250) — matrix operand order repaired and checked by reading, not compiled, not run on hardware
//
#include <hip/hip_runtime.h>
#include <hip/hip_bf16.h>
#include <math.h>

typedef __attribute__((ext_vector_type(16))) _Float16 v16h;
typedef __attribute__((ext_vector_type(8)))  _Float16 v8h;
typedef __attribute__((ext_vector_type(8)))  float    v8f;

#define SHUF16(a,b) __builtin_shufflevector((a),(b),0,1,2,3,4,5,6,7,8,9,10,11,12,13,14,15)
__device__ __forceinline__ float lrelu01(float x) { return x > 0.0f ? x : 0.01f * x; }

// A fragment: per-lane K-runs {kb..kb+7, kb+16..kb+23}, kb = k0 + (lane>=16)*8
__device__ __forceinline__ v16h load_afrag(const _Float16* p) {
    v8h c0 = *(const v8h*)p;
    v8h c1 = *(const v8h*)(p + 16);
    return SHUF16(c0, c1);
}
// B fragment: per-lane K-runs {kb..kb+7, kb+16..kb+23}, kb = k0 + (lane>=16)*8
__device__ __forceinline__ v16h load_bfrag(const _Float16* p) {
    v8h c0 = *(const v8h*)p;
    v8h c1 = *(const v8h*)(p + 16);
    return SHUF16(c0, c1);
}

// ---------------------------------------------------------------------------
// Gaussian magnitude encoding: (32,16,80,80) f32 -> (512,20,80,80) f16
// ---------------------------------------------------------------------------
__global__ __launch_bounds__(256) void enc_kernel(const float* __restrict__ batch,
                                                  _Float16* __restrict__ enc, int total)
{
    int idx = blockIdx.x * 256 + threadIdx.x;
    if (idx >= total) return;
    const float INV  = 2.5253813613805268f;  // 1/(sqrt(2)*0.28)
    const float NORM = 1.4247940270785044f;  // 1/(sqrt(2*pi)*0.28)
    int img = idx / (20 * 6400);
    int rem = idx - img * (20 * 6400);
    int c   = rem / 6400;
    int pix = rem - c * 6400;
    float x = batch[(size_t)img * 6400 + pix];
    float t = tanhf(x * (2.0f / 255.0f) - 1.0f);
    float mag = -1.0f + (2.0f / 19.0f) * (float)c;
    float u = (t + mag) * INV;
    enc[idx] = (_Float16)(NORM * expf(-u * u));
}

// ---------------------------------------------------------------------------
// Direct 3x3 stride-2 VALID conv + bias + lrelu. One block per (img, oc),
// weights staged in LDS, threads sweep spatial outputs.
// ---------------------------------------------------------------------------
__global__ __launch_bounds__(256) void conv3x3_s2(const _Float16* __restrict__ in,
                                                  const float* __restrict__ w,
                                                  const float* __restrict__ bias,
                                                  _Float16* __restrict__ out,
                                                  int IC, int IH, int IW,
                                                  int OC, int OH, int OW)
{
    __shared__ float wsm[288]; // IC*9 <= 288
    int img = blockIdx.x / OC;
    int oc  = blockIdx.x - img * OC;
    int nw  = IC * 9;
    for (int t = threadIdx.x; t < nw; t += blockDim.x) wsm[t] = w[(size_t)oc * nw + t];
    __syncthreads();
    float bv = bias[oc];
    const _Float16* ibase = in + (size_t)img * IC * IH * IW;
    _Float16* obase = out + ((size_t)img * OC + oc) * OH * OW;
    int npos = OH * OW;
    for (int pos = threadIdx.x; pos < npos; pos += blockDim.x) {
        int oy = pos / OW, ox = pos - oy * OW;
        float acc = bv;
        for (int ic = 0; ic < IC; ++ic) {
            const _Float16* ip = ibase + ((size_t)ic * IH + oy * 2) * IW + ox * 2;
            const float* wp = wsm + ic * 9;
#pragma unroll
            for (int ky = 0; ky < 3; ++ky)
#pragma unroll
                for (int kx = 0; kx < 3; ++kx)
                    acc += (float)ip[ky * IW + kx] * wp[ky * 3 + kx];
        }
        obase[pos] = (_Float16)lrelu01(acc);
    }
}

// ---------------------------------------------------------------------------
// Weight prep: W[K][N] f32 -> Bt[N][K] f16 (transposed, row-major N x K)
// ---------------------------------------------------------------------------
__global__ __launch_bounds__(256) void wcvt_kernel(const float* __restrict__ W,
                                                   _Float16* __restrict__ Bt,
                                                   int K, int N)
{
    int idx = blockIdx.x * 256 + threadIdx.x;
    if (idx >= K * N) return;
    int k = idx / N, n = idx - k * N;
    Bt[(size_t)n * K + k] = (_Float16)W[idx];
}

// ---------------------------------------------------------------------------
// WMMA f16 GEMM: C[M,N] = lrelu(A[M,K] @ Bt^T + bias). Bt is N x K (ldb=K).
// 256 threads = 8 waves; block tile 128x128; wave tile 32x64 (2x4 WMMA).
// Out-of-range rows/cols are CLAMPED at the load (garbage only reaches
// accumulator rows/cols that the guarded epilogue never stores), so the
// k-loop has zero branches. Software-pipelined with K-unroll of 2
// (K is always a multiple of 64 here) so WMMAs overlap fragment loads.
// ---------------------------------------------------------------------------
__global__ __launch_bounds__(256) void gemm_wmma(const _Float16* __restrict__ A, int lda,
                                                 const _Float16* __restrict__ Bt, int ldb,
                                                 const float* __restrict__ bias,
                                                 _Float16* __restrict__ Ch,
                                                 float* __restrict__ Cf, int ldc,
                                                 int M, int N, int K, int do_lrelu)
{
    const int lane = threadIdx.x & 31;
    const int wv   = threadIdx.x >> 5;
    const int m_base = blockIdx.y * 128 + (wv & 3) * 32;
    const int n_base = blockIdx.x * 128 + (wv >> 2) * 64;
    const int lr = lane & 15;
    const int lh = lane >> 4;

    // per-lane clamped base pointers (k-invariant)
    const _Float16* pa[2];
    const _Float16* pb[4];
#pragma unroll
    for (int i = 0; i < 2; ++i) {
        int row = m_base + i * 16 + lr;
        if (row > M - 1) row = M - 1;
        pa[i] = A + (size_t)row * lda + lh * 8;
    }
#pragma unroll
    for (int j = 0; j < 4; ++j) {
        int col = n_base + j * 16 + lr;
        if (col > N - 1) col = N - 1;
        pb[j] = Bt + (size_t)col * ldb + lh * 8;
    }

    v8f acc[2][4];
#pragma unroll
    for (int i = 0; i < 2; ++i)
#pragma unroll
        for (int j = 0; j < 4; ++j) acc[i][j] = {};

    v16h a0[2], b0[4], a1[2], b1[4];
#pragma unroll
    for (int i = 0; i < 2; ++i) a0[i] = load_afrag(pa[i]);
#pragma unroll
    for (int j = 0; j < 4; ++j) b0[j] = load_bfrag(pb[j]);

    for (int k0 = 0; k0 < K; k0 += 64) {
        // prefetch chunk k0+32 (always in range: K % 64 == 0)
#pragma unroll
        for (int i = 0; i < 2; ++i) a1[i] = load_afrag(pa[i] + k0 + 32);
#pragma unroll
        for (int j = 0; j < 4; ++j) b1[j] = load_bfrag(pb[j] + k0 + 32);

#pragma unroll
        for (int i = 0; i < 2; ++i)
#pragma unroll
            for (int j = 0; j < 4; ++j)
                acc[i][j] = __builtin_amdgcn_wmma_f32_16x16x32_f16(
                    false, a0[i], false, b0[j], (short)0, acc[i][j], false, false);

        if (k0 + 64 < K) {
#pragma unroll
            for (int i = 0; i < 2; ++i) a0[i] = load_afrag(pa[i] + k0 + 64);
#pragma unroll
            for (int j = 0; j < 4; ++j) b0[j] = load_bfrag(pb[j] + k0 + 64);
        }

#pragma unroll
        for (int i = 0; i < 2; ++i)
#pragma unroll
            for (int j = 0; j < 4; ++j)
                acc[i][j] = __builtin_amdgcn_wmma_f32_16x16x32_f16(
                    false, a1[i], false, b1[j], (short)0, acc[i][j], false, false);
    }

#pragma unroll
    for (int i = 0; i < 2; ++i) {
#pragma unroll
        for (int j = 0; j < 4; ++j) {
            int col = n_base + j * 16 + lr;
            if (col >= N) continue;
            float bv = bias ? bias[col] : 0.0f;
#pragma unroll
            for (int v = 0; v < 8; ++v) {
                int row = m_base + i * 16 + lh * 8 + v;
                if (row >= M) continue;
                float x = acc[i][j][v] + bv;
                if (do_lrelu) x = lrelu01(x);
                if (Cf) Cf[(size_t)row * ldc + col] = x;
                else    Ch[(size_t)row * ldc + col] = (_Float16)x;
            }
        }
    }
}

// ---------------------------------------------------------------------------
// Build g-stage input: rows = 256*81, cols = 512.
// row r: q=r/81 (=bb*8+o), a=(r%81)/9, b=r%9; [obj(a)(256) | obj(b)(256)]
// obj(i) = [panel[bb, i<8?i:8+o][0:247] | onehot9(i)]
// panel stored f16 with row stride 256.
// ---------------------------------------------------------------------------
__global__ __launch_bounds__(256) void build_g_kernel(const _Float16* __restrict__ panel,
                                                      _Float16* __restrict__ out, int total)
{
    int idx = blockIdx.x * 256 + threadIdx.x;
    if (idx >= total) return;
    int r = idx >> 9;            // /512
    int c = idx & 511;
    int q = r / 81;
    int pr = r - q * 81;
    int a = pr / 9, b = pr - (pr / 9) * 9;
    int obj = (c < 256) ? a : b;
    int cc = c & 255;
    int bb = q >> 3, o = q & 7;
    _Float16 val;
    if (cc < 247) {
        int pidx = (obj < 8) ? obj : (8 + o);
        val = panel[((size_t)(bb * 16 + pidx)) * 256 + cc];
    } else {
        val = (_Float16)((cc - 247 == obj) ? 1.0f : 0.0f);
    }
    out[idx] = val;
}

// ---------------------------------------------------------------------------
// Build h-stage input from sum_j f32 (2304 x 256): rows = 256*81, cols = 512
// ---------------------------------------------------------------------------
__global__ __launch_bounds__(256) void build_h_kernel(const float* __restrict__ sumj,
                                                      _Float16* __restrict__ out, int total)
{
    int idx = blockIdx.x * 256 + threadIdx.x;
    if (idx >= total) return;
    int r = idx >> 9;
    int c = idx & 511;
    int q = r / 81;
    int pr = r - q * 81;
    int a = pr / 9, b = pr - (pr / 9) * 9;
    int obj = (c < 256) ? a : b;
    int cc = c & 255;
    out[idx] = (_Float16)sumj[((size_t)(q * 9 + obj)) * 256 + cc];
}

// ---------------------------------------------------------------------------
// sum over pair-index b: z (20736 x 256 f16) -> sum_j (2304 x 256 f32)
// ---------------------------------------------------------------------------
__global__ __launch_bounds__(256) void sum9_kernel(const _Float16* __restrict__ z,
                                                   float* __restrict__ sumj, int total)
{
    int idx = blockIdx.x * 256 + threadIdx.x;
    if (idx >= total) return;
    int row = idx >> 8;          // q*9 + a
    int c   = idx & 255;
    int q = row / 9, a = row - q * 9;
    size_t base = ((size_t)(q * 81 + a * 9)) * 256 + c;
    float s = 0.0f;
#pragma unroll
    for (int b = 0; b < 9; ++b) s += (float)z[base + (size_t)b * 256];
    sumj[idx] = s;
}

// ---------------------------------------------------------------------------
// h_sum: sum over a of sum_j -> (256 x 256) f32 and f16
// ---------------------------------------------------------------------------
__global__ __launch_bounds__(256) void hsum_kernel(const float* __restrict__ sumj,
                                                   float* __restrict__ hs,
                                                   _Float16* __restrict__ hs_h, int total)
{
    int idx = blockIdx.x * 256 + threadIdx.x;
    if (idx >= total) return;
    int q = idx >> 8;
    int c = idx & 255;
    float s = 0.0f;
#pragma unroll
    for (int a = 0; a < 9; ++a) s += sumj[((size_t)(q * 9 + a)) * 256 + c];
    hs[idx] = s;
    hs_h[idx] = (_Float16)s;
}

// ---------------------------------------------------------------------------
// loss = mean(h_sum^2) + mean(result^2); one block of 256 threads.
// ---------------------------------------------------------------------------
__global__ __launch_bounds__(256) void loss_kernel(const float* __restrict__ hs,
                                                   const float* __restrict__ result,
                                                   float* __restrict__ out_loss)
{
    __shared__ float red[256];
    int tid = threadIdx.x;
    float s = 0.0f;
    for (int i = tid; i < 65536; i += 256) { float v = hs[i]; s += v * v; }
    s *= (1.0f / 65536.0f);
    float rv = result[tid];
    s += rv * rv * (1.0f / 256.0f);
    red[tid] = s;
    __syncthreads();
    for (int off = 128; off > 0; off >>= 1) {
        if (tid < off) red[tid] += red[tid + off];
        __syncthreads();
    }
    if (tid == 0) *out_loss = red[0];
}

// ---------------------------------------------------------------------------
// Host-side orchestration
// ---------------------------------------------------------------------------
static inline size_t align_up(size_t x) { return (x + 255) & ~(size_t)255; }

extern "C" void kernel_launch(void* const* d_in, const int* in_sizes, int n_in,
                              void* d_out, int out_size, void* d_ws, size_t ws_size,
                              hipStream_t stream)
{
    (void)in_sizes; (void)n_in; (void)out_size; (void)ws_size;

    const float* batch   = (const float*)d_in[0];
    const float* conv_w1 = (const float*)d_in[1];  const float* conv_b1 = (const float*)d_in[2];
    const float* conv_w2 = (const float*)d_in[3];  const float* conv_b2 = (const float*)d_in[4];
    const float* conv_w3 = (const float*)d_in[5];  const float* conv_b3 = (const float*)d_in[6];
    const float* conv_w4 = (const float*)d_in[7];  const float* conv_b4 = (const float*)d_in[8];
    const float* post_w  = (const float*)d_in[9];  const float* post_b  = (const float*)d_in[10];
    const float* g_w1 = (const float*)d_in[11]; const float* g_b1 = (const float*)d_in[12];
    const float* g_w2 = (const float*)d_in[13]; const float* g_b2 = (const float*)d_in[14];
    const float* g_w3 = (const float*)d_in[15]; const float* g_b3 = (const float*)d_in[16];
    const float* g_w4 = (const float*)d_in[17]; const float* g_b4 = (const float*)d_in[18];
    const float* h_w1 = (const float*)d_in[19]; const float* h_b1 = (const float*)d_in[20];
    const float* h_w2 = (const float*)d_in[21]; const float* h_b2 = (const float*)d_in[22];
    const float* h_w3 = (const float*)d_in[23]; const float* h_b3 = (const float*)d_in[24];
    const float* f_w1 = (const float*)d_in[25]; const float* f_b1 = (const float*)d_in[26];
    const float* f_w2 = (const float*)d_in[27]; const float* f_b2 = (const float*)d_in[28];
    const float* ff_w = (const float*)d_in[29]; const float* ff_b = (const float*)d_in[30];
    float* out = (float*)d_out;  // [0..255] answer, [256] loss

    // workspace carve-out
    char* base = (char*)d_ws;
    size_t off = 0;
    auto alloc = [&](size_t bytes) -> void* { void* p = base + off; off = align_up(off + bytes); return p; };

    _Float16* enc   = (_Float16*)alloc((size_t)512 * 20 * 6400 * 2);
    _Float16* c1o   = (_Float16*)alloc((size_t)512 * 32 * 39 * 39 * 2);
    _Float16* c2o   = (_Float16*)alloc((size_t)512 * 32 * 19 * 19 * 2);
    _Float16* c3o   = (_Float16*)alloc((size_t)512 * 32 * 9 * 9 * 2);
    _Float16* feat  = (_Float16*)alloc((size_t)512 * 512 * 2);      // conv4 out == post A
    _Float16* panel = (_Float16*)alloc((size_t)512 * 256 * 2);      // ldc=256, cols 0..246 valid
    _Float16* buf0  = (_Float16*)alloc((size_t)20736 * 512 * 2);
    _Float16* buf1  = (_Float16*)alloc((size_t)20736 * 512 * 2);
    float*    sumj  = (float*)   alloc((size_t)2304 * 256 * 4);
    float*    hsf   = (float*)   alloc((size_t)256 * 256 * 4);
    _Float16* hsh   = (_Float16*)alloc((size_t)256 * 256 * 2);
    _Float16* fb1   = (_Float16*)alloc((size_t)256 * 256 * 2);
    _Float16* fb2   = (_Float16*)alloc((size_t)256 * 256 * 2);
    _Float16* postT = (_Float16*)alloc((size_t)247 * 512 * 2);
    _Float16* g1T   = (_Float16*)alloc((size_t)512 * 512 * 2);
    _Float16* g2T   = (_Float16*)alloc((size_t)512 * 512 * 2);
    _Float16* g3T   = (_Float16*)alloc((size_t)512 * 512 * 2);
    _Float16* g4T   = (_Float16*)alloc((size_t)256 * 512 * 2);
    _Float16* h1T   = (_Float16*)alloc((size_t)3 * 256 * 512 * 2);
    _Float16* h2T   = (_Float16*)alloc((size_t)3 * 256 * 256 * 2);
    _Float16* h3T   = (_Float16*)alloc((size_t)3 * 256 * 256 * 2);
    _Float16* f1T   = (_Float16*)alloc((size_t)256 * 256 * 2);
    _Float16* f2T   = (_Float16*)alloc((size_t)256 * 256 * 2);
    _Float16* ffT   = (_Float16*)alloc((size_t)1 * 256 * 2);

    auto grid1 = [](int n) { return dim3((unsigned)((n + 255) / 256)); };

    // ---- weight transposition to f16 Bt[N][K] ----
    wcvt_kernel<<<grid1(512 * 247), 256, 0, stream>>>(post_w, postT, 512, 247);
    wcvt_kernel<<<grid1(512 * 512), 256, 0, stream>>>(g_w1, g1T, 512, 512);
    wcvt_kernel<<<grid1(512 * 512), 256, 0, stream>>>(g_w2, g2T, 512, 512);
    wcvt_kernel<<<grid1(512 * 512), 256, 0, stream>>>(g_w3, g3T, 512, 512);
    wcvt_kernel<<<grid1(512 * 256), 256, 0, stream>>>(g_w4, g4T, 512, 256);
    for (int i = 0; i < 3; ++i) {
        wcvt_kernel<<<grid1(512 * 256), 256, 0, stream>>>(h_w1 + (size_t)i * 512 * 256, h1T + (size_t)i * 256 * 512, 512, 256);
        wcvt_kernel<<<grid1(256 * 256), 256, 0, stream>>>(h_w2 + (size_t)i * 256 * 256, h2T + (size_t)i * 256 * 256, 256, 256);
        wcvt_kernel<<<grid1(256 * 256), 256, 0, stream>>>(h_w3 + (size_t)i * 256 * 256, h3T + (size_t)i * 256 * 256, 256, 256);
    }
    wcvt_kernel<<<grid1(256 * 256), 256, 0, stream>>>(f_w1, f1T, 256, 256);
    wcvt_kernel<<<grid1(256 * 256), 256, 0, stream>>>(f_w2, f2T, 256, 256);
    wcvt_kernel<<<grid1(256 * 1),   256, 0, stream>>>(ff_w, ffT, 256, 1);

    // ---- encoding + convs ----
    enc_kernel<<<grid1(512 * 20 * 6400), 256, 0, stream>>>(batch, enc, 512 * 20 * 6400);
    conv3x3_s2<<<dim3(512 * 32), 256, 0, stream>>>(enc, conv_w1, conv_b1, c1o, 20, 80, 80, 32, 39, 39);
    conv3x3_s2<<<dim3(512 * 32), 256, 0, stream>>>(c1o, conv_w2, conv_b2, c2o, 32, 39, 39, 32, 19, 19);
    conv3x3_s2<<<dim3(512 * 32), 256, 0, stream>>>(c2o, conv_w3, conv_b3, c3o, 32, 19, 19, 32, 9, 9);
    conv3x3_s2<<<dim3(512 * 32), 256, 0, stream>>>(c3o, conv_w4, conv_b4, feat, 32, 9, 9, 32, 4, 4);

    auto gemm_grid = [](int M, int N) { return dim3((unsigned)((N + 127) / 128), (unsigned)((M + 127) / 128)); };

    // ---- post projection (no activation) ----
    gemm_wmma<<<gemm_grid(512, 247), 256, 0, stream>>>(feat, 512, postT, 512, post_b,
                                                       panel, nullptr, 256, 512, 247, 512, 0);

    // ---- g MLP on 20736 x 512 ----
    const int RC = 20736, TOT = RC * 512;
    build_g_kernel<<<grid1(TOT), 256, 0, stream>>>(panel, buf0, TOT);
    gemm_wmma<<<gemm_grid(RC, 512), 256, 0, stream>>>(buf0, 512, g1T, 512, g_b1, buf1, nullptr, 512, RC, 512, 512, 1);
    gemm_wmma<<<gemm_grid(RC, 512), 256, 0, stream>>>(buf1, 512, g2T, 512, g_b2, buf0, nullptr, 512, RC, 512, 512, 1);
    gemm_wmma<<<gemm_grid(RC, 512), 256, 0, stream>>>(buf0, 512, g3T, 512, g_b3, buf1, nullptr, 512, RC, 512, 512, 1);
    gemm_wmma<<<gemm_grid(RC, 256), 256, 0, stream>>>(buf1, 512, g4T, 512, g_b4, buf0, nullptr, 256, RC, 256, 512, 1);
    sum9_kernel<<<grid1(2304 * 256), 256, 0, stream>>>(buf0, sumj, 2304 * 256);

    // ---- h MLP iterations ----
    for (int i = 0; i < 3; ++i) {
        build_h_kernel<<<grid1(TOT), 256, 0, stream>>>(sumj, buf1, TOT);
        gemm_wmma<<<gemm_grid(RC, 256), 256, 0, stream>>>(buf1, 512, h1T + (size_t)i * 256 * 512, 512,
                                                          h_b1 + (size_t)i * 256, buf0, nullptr, 256, RC, 256, 512, 1);
        gemm_wmma<<<gemm_grid(RC, 256), 256, 0, stream>>>(buf0, 256, h2T + (size_t)i * 256 * 256, 256,
                                                          h_b2 + (size_t)i * 256, buf1, nullptr, 256, RC, 256, 256, 1);
        gemm_wmma<<<gemm_grid(RC, 256), 256, 0, stream>>>(buf1, 256, h3T + (size_t)i * 256 * 256, 256,
                                                          h_b3 + (size_t)i * 256, buf0, nullptr, 256, RC, 256, 256, 1);
        sum9_kernel<<<grid1(2304 * 256), 256, 0, stream>>>(buf0, sumj, 2304 * 256);
    }

    // ---- h_sum, f MLP, answer, loss ----
    hsum_kernel<<<grid1(256 * 256), 256, 0, stream>>>(sumj, hsf, hsh, 256 * 256);
    gemm_wmma<<<gemm_grid(256, 256), 256, 0, stream>>>(hsh, 256, f1T, 256, f_b1, fb1, nullptr, 256, 256, 256, 256, 1);
    gemm_wmma<<<gemm_grid(256, 256), 256, 0, stream>>>(fb1, 256, f2T, 256, f_b2, fb2, nullptr, 256, 256, 256, 256, 1);
    gemm_wmma<<<gemm_grid(256, 1),   256, 0, stream>>>(fb2, 256, ffT, 256, ff_b, nullptr, out, 1, 256, 1, 256, 0);
    loss_kernel<<<dim3(1), 256, 0, stream>>>(hsf, out, out + 256);
}
